// NetNode_87866440941572
// MI455X (gfx1250) — matrix operand order repaired and checked by reading, not compiled, not run on hardware
//
#include <hip/hip_runtime.h>
#include <hip/hip_bf16.h>
#include <math.h>

typedef __attribute__((ext_vector_type(16))) _Float16 v16h;
typedef __attribute__((ext_vector_type(8)))  float    v8f;

#define B_   256
#define N_   512
#define E_   2048
#define EMB_ 64
#define BN_  (B_ * N_)     // 131072
#define BE_  (B_ * E_)     // 524288
#define C_   128

static inline int cdiv(int a, int b) { return (a + b - 1) / b; }

// ---------------------------------------------------------------------------
// WMMA fragment loaders (layouts per CDNA5 ISA §7.12.2, wave32)
// ---------------------------------------------------------------------------

// A: f32 row-major [M x K]; produce 16x32 f16 A-fragment for rows tM..tM+15,
// K-slice k0..k0+31.  lanes 0-15: K (k0+0..7, k0+16..23); lanes 16-31: +8.
__device__ __forceinline__ v16h load_a_frag(const float* __restrict__ A,
                                            int K, int row, int k0, int lane) {
  const float* p = A + (size_t)row * K + k0 + ((lane & 16) ? 8 : 0);
  v16h a;
#pragma unroll
  for (int h = 0; h < 8; ++h)  a[h] = (_Float16)p[h];
#pragma unroll
  for (int h = 8; h < 16; ++h) a[h] = (_Float16)p[h + 8];
  return a;
}

// Wt: f16 column-major [Nc x K]  (Wt[n*K+k] == W[k][n]); produce 32x16
// B-fragment for cols n0..n0+15, K-slice k0..k0+31.
// lanes 0-15: K (k0+0..7, k0+16..23); lanes 16-31: +8; col = n0 + (lane&15).
__device__ __forceinline__ v16h load_b_frag(const _Float16* __restrict__ Wt,
                                            int K, int col, int k0, int lane) {
  const _Float16* p = Wt + (size_t)col * K + k0 + ((lane & 16) ? 8 : 0);
  v16h b;
#pragma unroll
  for (int h = 0; h < 8; ++h)  b[h] = p[h];
#pragma unroll
  for (int h = 8; h < 16; ++h) b[h] = p[h + 8];
  return b;
}

// C[M x (NT*16)] = (RELU?)( A1@W1 (+ A2@W2 if DUAL) + bias ), f32 out/accum.
// One wave owns a 16-row band across ALL NT column tiles: each A fragment is
// loaded once per K-step and feeds NT WMMAs (A-side traffic read exactly once
// per GEMM; weight fragments come from the hot WGP$/L2 working set).
// Requires M % 16 == 0, K % 32 == 0.
template<int NT, bool DUAL, bool RELU>
__global__ __launch_bounds__(256)
void wmma_gemm_kernel(const float* __restrict__ A1, const float* __restrict__ A2,
                      const _Float16* __restrict__ W1t, const _Float16* __restrict__ W2t,
                      const float* __restrict__ bias, float* __restrict__ C,
                      int M, int K) {
  const int Nc  = NT * 16;
  int gtid = blockIdx.x * blockDim.x + threadIdx.x;
  int wave = gtid >> 5;
  int lane = threadIdx.x & 31;
  if (wave >= (M >> 4)) return;              // uniform per-wave branch
  int tM    = wave << 4;
  int arow  = tM + (lane & 15);
  int bcol0 = lane & 15;

  v8f acc[NT];
#pragma unroll
  for (int nt = 0; nt < NT; ++nt) acc[nt] = (v8f){};

  for (int k0 = 0; k0 < K; k0 += 32) {
    v16h a = load_a_frag(A1, K, arow, k0, lane);
#pragma unroll
    for (int nt = 0; nt < NT; ++nt) {
      v16h b = load_b_frag(W1t, K, nt * 16 + bcol0, k0, lane);
      acc[nt] = __builtin_amdgcn_wmma_f32_16x16x32_f16(false, a, false, b,
                                                       (short)0, acc[nt], false, false);
    }
    if (DUAL) {
      v16h a2 = load_a_frag(A2, K, arow, k0, lane);
#pragma unroll
      for (int nt = 0; nt < NT; ++nt) {
        v16h b = load_b_frag(W2t, K, nt * 16 + bcol0, k0, lane);
        acc[nt] = __builtin_amdgcn_wmma_f32_16x16x32_f16(false, a2, false, b,
                                                         (short)0, acc[nt], false, false);
      }
    }
  }

  // C layout: VGPR r -> row tM + r + (lane>=16 ? 8 : 0), col = n0 + (lane&15)
  int rbase = tM + ((lane & 16) ? 8 : 0);
#pragma unroll
  for (int nt = 0; nt < NT; ++nt) {
    int   col = nt * 16 + bcol0;
    float bv  = bias[col];
#pragma unroll
    for (int r = 0; r < 8; ++r) {
      float v = acc[nt][r] + bv;
      if (RELU) v = fmaxf(v, 0.0f);
      C[(size_t)(rbase + r) * Nc + col] = v;
    }
  }
}

// ---------------------------------------------------------------------------
// Supporting kernels
// ---------------------------------------------------------------------------

__global__ void zero_kernel(float* __restrict__ p, size_t n) {
  size_t i = (size_t)blockIdx.x * blockDim.x + threadIdx.x;
  if (i < n) p[i] = 0.0f;
}

// f32 row-major [K x Nc] -> f16 column-major [Nc x K]
__global__ void convert_wt_kernel(const float* __restrict__ W, _Float16* __restrict__ Wt,
                                  int K, int Nc) {
  int i = blockIdx.x * blockDim.x + threadIdx.x;
  if (i >= K * Nc) return;
  int k = i / Nc, n = i % Nc;
  Wt[(size_t)n * K + k] = (_Float16)W[i];
}

// x[i] = concat(emb_item[item_id[i]], emb_cat[cat_id[i]])  (one block per node)
__global__ void embed_kernel(const int* __restrict__ item_id, const int* __restrict__ cat_id,
                             const float* __restrict__ ei, const float* __restrict__ ec,
                             float* __restrict__ x) {
  int i = blockIdx.x;
  int c = threadIdx.x;  // blockDim = 128
  float v = (c < EMB_) ? ei[(size_t)item_id[i] * EMB_ + c]
                       : ec[(size_t)cat_id[i] * EMB_ + (c - EMB_)];
  x[(size_t)i * C_ + c] = v;
}

__global__ void init_edges_kernel(const int* __restrict__ s, const int* __restrict__ d,
                                  int* __restrict__ sw, int* __restrict__ dw,
                                  int* __restrict__ mw, int nE) {
  int e = blockIdx.x * blockDim.x + threadIdx.x;
  if (e >= nE) return;
  sw[e] = s[e]; dw[e] = d[e]; mw[e] = 1;
}

// agg[dst[e]] += x[src[e]]  (masked); one block per edge, one thread per channel
__global__ void scatter_add_kernel(const float* __restrict__ x,
                                   const int* __restrict__ src, const int* __restrict__ dst,
                                   const int* __restrict__ mask, float* __restrict__ agg) {
  int e = blockIdx.x;
  int c = threadIdx.x;  // blockDim = 128
  if (mask[e]) {
    atomicAdd(&agg[(size_t)dst[e] * C_ + c], x[(size_t)src[e] * C_ + c]);
  }
}

__global__ void score_kernel(const float* __restrict__ x, const float* __restrict__ p,
                             float* __restrict__ score, int Mn) {
  int i = blockIdx.x * blockDim.x + threadIdx.x;
  if (i >= Mn) return;
  float s = 0.f, pn = 0.f;
  for (int c = 0; c < C_; ++c) { s += x[(size_t)i * C_ + c] * p[c]; pn += p[c] * p[c]; }
  score[i] = s * rsqrtf(pn);
}

// One block per graph. Stable descending rank over scores (ties -> lower index),
// matching jax.lax.top_k ordering; keep rank < k, gate with tanh(score).
__global__ void topk_kernel(const float* __restrict__ x, const float* __restrict__ score,
                            float* __restrict__ xk, int* __restrict__ mapping,
                            int n, int k) {
  __shared__ float s[512];
  int b = blockIdx.x;
  int t = threadIdx.x;  // blockDim = 512
  if (t < n) s[t] = score[(size_t)b * n + t];
  __syncthreads();
  if (t >= n) return;
  int   gi = b * n + t;
  float st = s[t];
  int rank = 0;
  for (int j = 0; j < n; ++j) {
    float sj = s[j];
    rank += (sj > st) || (sj == st && j < t);
  }
  if (rank < k) {
    int ni = b * k + rank;
    mapping[gi] = ni;
    float gate = tanhf(st);
    for (int c = 0; c < C_; ++c)
      xk[(size_t)ni * C_ + c] = x[(size_t)gi * C_ + c] * gate;
  } else {
    mapping[gi] = -1;
  }
}

__global__ void remap_edges_kernel(int* __restrict__ src, int* __restrict__ dst,
                                   int* __restrict__ mask, const int* __restrict__ mapping,
                                   int nE) {
  int e = blockIdx.x * blockDim.x + threadIdx.x;
  if (e >= nE) return;
  int ns = mapping[src[e]];
  int nd = mapping[dst[e]];
  int ok = (ns >= 0) && (nd >= 0) && mask[e];
  src[e]  = ns >= 0 ? ns : 0;
  dst[e]  = nd >= 0 ? nd : 0;
  mask[e] = ok;
}

// gsum[b] (+)= concat(max over k, mean over k); one block per graph, thread per channel
__global__ void readout_kernel(const float* __restrict__ xk, float* __restrict__ gsum,
                               int k, int first) {
  int b = blockIdx.x;
  int c = threadIdx.x;  // blockDim = 128
  float mx = -INFINITY, sm = 0.f;
  for (int r = 0; r < k; ++r) {
    float v = xk[((size_t)b * k + r) * C_ + c];
    mx = fmaxf(mx, v);
    sm += v;
  }
  float* g = gsum + (size_t)b * (2 * C_);
  float mean = sm / (float)k;
  if (first) { g[c] = mx; g[C_ + c] = mean; }
  else       { g[c] += mx; g[C_ + c] += mean; }
}

// BatchNorm (training-mode biased stats over batch of 256) + ReLU, in place.
// One block per column, blockDim = 256 (== batch).
__global__ void bn_relu_kernel(float* __restrict__ h, const float* __restrict__ gamma,
                               const float* __restrict__ beta, int Cc) {
  __shared__ float red[256];
  int c = blockIdx.x;
  int t = threadIdx.x;
  float v = h[(size_t)t * Cc + c];
  red[t] = v;
  __syncthreads();
  for (int s = 128; s > 0; s >>= 1) { if (t < s) red[t] += red[t + s]; __syncthreads(); }
  float mean = red[0] / 256.0f;
  __syncthreads();
  float d = v - mean;
  red[t] = d * d;
  __syncthreads();
  for (int s = 128; s > 0; s >>= 1) { if (t < s) red[t] += red[t + s]; __syncthreads(); }
  float var = red[0] / 256.0f;
  float o = gamma[c] * d * rsqrtf(var + 1e-5f) + beta[c];
  h[(size_t)t * Cc + c] = fmaxf(o, 0.0f);
}

// out[i] = sigmoid(<emb_item[item_id[i]], g[i / N]>)
__global__ void final_kernel(const int* __restrict__ item_id, const float* __restrict__ emb_item,
                             const float* __restrict__ g, float* __restrict__ out) {
  int i = blockIdx.x * blockDim.x + threadIdx.x;
  if (i >= BN_) return;
  int b = i / N_;
  const float* e  = emb_item + (size_t)item_id[i] * EMB_;
  const float* gb = g + (size_t)b * EMB_;
  float s = 0.f;
#pragma unroll
  for (int d = 0; d < EMB_; ++d) s += e[d] * gb[d];
  out[i] = 1.0f / (1.0f + expf(-s));
}

// ---------------------------------------------------------------------------
// Host orchestration
// ---------------------------------------------------------------------------

extern "C" void kernel_launch(void* const* d_in, const int* in_sizes, int n_in,
                              void* d_out, int out_size, void* d_ws, size_t ws_size,
                              hipStream_t stream) {
  const int*   item_id  = (const int*)d_in[0];
  const int*   cat_id   = (const int*)d_in[1];
  const int*   src_in   = (const int*)d_in[2];
  const int*   dst_in   = (const int*)d_in[3];
  const float* emb_item = (const float*)d_in[4];
  const float* emb_cat  = (const float*)d_in[5];
  const float* Wr[3] = {(const float*)d_in[6],  (const float*)d_in[10], (const float*)d_in[14]};
  const float* Wn[3] = {(const float*)d_in[7],  (const float*)d_in[11], (const float*)d_in[15]};
  const float* bb[3] = {(const float*)d_in[8],  (const float*)d_in[12], (const float*)d_in[16]};
  const float* pp[3] = {(const float*)d_in[9],  (const float*)d_in[13], (const float*)d_in[17]};
  const float* fc1_W = (const float*)d_in[18];
  const float* fc1_b = (const float*)d_in[19];
  const float* bn1_g = (const float*)d_in[20];
  const float* bn1_b = (const float*)d_in[21];
  const float* fc2_W = (const float*)d_in[22];
  const float* fc2_b = (const float*)d_in[23];
  const float* bn2_g = (const float*)d_in[24];
  const float* bn2_b = (const float*)d_in[25];
  float* out = (float*)d_out;

  // bump allocator over d_ws
  char* wptr = (char*)d_ws;
  auto alloc = [&](size_t bytes) -> void* {
    void* p = (void*)wptr;
    wptr += (bytes + 255) & ~(size_t)255;
    return p;
  };
  float* xA   = (float*)alloc((size_t)BN_ * C_ * 4);   // node features / pooled out
  float* xB   = (float*)alloc((size_t)BN_ * C_ * 4);   // aggregation buffer
  float* xC   = (float*)alloc((size_t)BN_ * C_ * 4);   // conv output
  int*   srcW = (int*)  alloc((size_t)BE_ * 4);
  int*   dstW = (int*)  alloc((size_t)BE_ * 4);
  int*   mskW = (int*)  alloc((size_t)BE_ * 4);
  int*   mapW = (int*)  alloc((size_t)BN_ * 4);
  float* scoreW = (float*)alloc((size_t)BN_ * 4);
  float* gsum = (float*)alloc((size_t)B_ * 2 * C_ * 4);
  float* h1   = (float*)alloc((size_t)B_ * 128 * 4);
  float* h2   = (float*)alloc((size_t)B_ * 64 * 4);
  _Float16* WrT[3], *WnT[3];
  for (int l = 0; l < 3; ++l) {
    WrT[l] = (_Float16*)alloc((size_t)128 * 128 * 2);
    WnT[l] = (_Float16*)alloc((size_t)128 * 128 * 2);
  }
  _Float16* fc1T = (_Float16*)alloc((size_t)256 * 128 * 2);
  _Float16* fc2T = (_Float16*)alloc((size_t)128 * 64 * 2);

  // 1. convert weights to f16 column-major
  for (int l = 0; l < 3; ++l) {
    convert_wt_kernel<<<cdiv(128 * 128, 256), 256, 0, stream>>>(Wr[l], WrT[l], 128, 128);
    convert_wt_kernel<<<cdiv(128 * 128, 256), 256, 0, stream>>>(Wn[l], WnT[l], 128, 128);
  }
  convert_wt_kernel<<<cdiv(256 * 128, 256), 256, 0, stream>>>(fc1_W, fc1T, 256, 128);
  convert_wt_kernel<<<cdiv(128 * 64, 256), 256, 0, stream>>>(fc2_W, fc2T, 128, 64);

  // 2. embedding gather + concat
  embed_kernel<<<BN_, 128, 0, stream>>>(item_id, cat_id, emb_item, emb_cat, xA);

  // 3. edge working copies
  init_edges_kernel<<<cdiv(BE_, 256), 256, 0, stream>>>(src_in, dst_in, srcW, dstW, mskW, BE_);

  // 4. three GraphConv + TopKPool + readout layers
  const int kArr[3] = {461, 415, 374};  // ceil(0.9*512), ceil(0.9*461), ceil(0.9*415)
  int nPrev = N_;
  for (int l = 0; l < 3; ++l) {
    int Mn = B_ * nPrev;
    int kL = kArr[l];
    // agg = 0 ; agg[dst] += x[src]
    zero_kernel<<<cdiv(Mn * C_, 256), 256, 0, stream>>>(xB, (size_t)Mn * C_);
    scatter_add_kernel<<<BE_, C_, 0, stream>>>(xA, srcW, dstW, mskW, xB);
    // xC = relu(x @ Wr + agg @ Wn + b)   [WMMA f16->f32, one wave per 16-row band]
    {
      int waves = Mn >> 4;
      wmma_gemm_kernel<8, true, true><<<cdiv(waves * 32, 256), 256, 0, stream>>>(
          xA, xB, WrT[l], WnT[l], bb[l], xC, Mn, C_);
    }
    // top-k pool: score, select, gate -> xA ; remap edges
    score_kernel<<<cdiv(Mn, 256), 256, 0, stream>>>(xC, pp[l], scoreW, Mn);
    topk_kernel<<<B_, 512, 0, stream>>>(xC, scoreW, xA, mapW, nPrev, kL);
    remap_edges_kernel<<<cdiv(BE_, 256), 256, 0, stream>>>(srcW, dstW, mskW, mapW, BE_);
    // readout accumulate
    readout_kernel<<<B_, C_, 0, stream>>>(xA, gsum, kL, l == 0 ? 1 : 0);
    nPrev = kL;
  }

  // 5. MLP head: fc1 + BN + ReLU, fc2 + BN + ReLU (extra ReLU is identity)
  {
    int waves = B_ >> 4;
    wmma_gemm_kernel<8, false, false><<<cdiv(waves * 32, 256), 256, 0, stream>>>(
        gsum, nullptr, fc1T, nullptr, fc1_b, h1, B_, 256);
  }
  bn_relu_kernel<<<128, 256, 0, stream>>>(h1, bn1_g, bn1_b, 128);
  {
    int waves = B_ >> 4;
    wmma_gemm_kernel<4, false, false><<<cdiv(waves * 32, 256), 256, 0, stream>>>(
        h1, nullptr, fc2T, nullptr, fc2_b, h2, B_, 128);
  }
  bn_relu_kernel<<<64, 256, 0, stream>>>(h2, bn2_g, bn2_b, 64);

  // 6. out = sigmoid(einsum('bnd,bd->bn', e_item, g))
  final_kernel<<<cdiv(BN_, 256), 256, 0, stream>>>(item_id, emb_item, h2, out);
}
